// MultiHeadAttentionLayer_53034256171290
// MI455X (gfx1250) — matrix operand order repaired and checked by reading, not compiled, not run on hardware
//
#include <hip/hip_runtime.h>
#include <math.h>

typedef __attribute__((ext_vector_type(8)))  float  v8f;
typedef __attribute__((ext_vector_type(8)))  __bf16 v8bf;
typedef __attribute__((ext_vector_type(16))) __bf16 v16bf;

#define BB 2
#define NN 8192
#define CC 128
#define HH 8
#define DD 32
#define KNN 9
#define NSPLIT 8

// ---------------------------------------------------------------------------
// Stage 1: projections + rotary + ||h||^2 + bf16 hi/lo split of h.
// One 256-thread block per (b,n).
// ---------------------------------------------------------------------------
__global__ __launch_bounds__(256) void mha_proj_rope(
    const float* __restrict__ h, const float* __restrict__ pos,
    const float* __restrict__ WQ, const float* __restrict__ WK,
    const float* __restrict__ WV,
    float* __restrict__ Qo, float* __restrict__ Ko, float* __restrict__ Vo,
    float* __restrict__ sqo, __bf16* __restrict__ hhi, __bf16* __restrict__ hlo) {
  __shared__ float sh_h[CC];
  __shared__ float sh_q[256];
  __shared__ float sh_k[256];
  __shared__ float s_sq;
  const int n = blockIdx.x;
  const int b = blockIdx.y;
  const int tid = threadIdx.x;
  const size_t row = (size_t)b * NN + n;

  if (tid == 0) s_sq = 0.0f;
  if (tid < CC) sh_h[tid] = h[row * CC + tid];
  __syncthreads();

  // bf16 two-term split of the h row (for the WMMA distance matmul)
  if (tid < CC) {
    const float x = sh_h[tid];
    const __bf16 hi = (__bf16)x;
    const float lo = x - (float)hi;
    hhi[row * CC + tid] = hi;
    hlo[row * CC + tid] = (__bf16)lo;
  }

  // ||h||^2 via per-wave butterfly + shared atomic
  float p = (tid < CC) ? sh_h[tid] * sh_h[tid] : 0.0f;
  #pragma unroll
  for (int off = 16; off > 0; off >>= 1) p += __shfl_xor(p, off, 32);
  if ((tid & 31) == 0) atomicAdd(&s_sq, p);

  // projections: each thread owns one of the 256 output columns
  float qv = 0.0f, kv = 0.0f, vv = 0.0f;
  #pragma unroll 8
  for (int j = 0; j < CC; ++j) {
    const float hv = sh_h[j];
    qv = fmaf(hv, WQ[j * 256 + tid], qv);
    kv = fmaf(hv, WK[j * 256 + tid], kv);
    vv = fmaf(hv, WV[j * 256 + tid], vv);
  }
  Vo[row * 256 + tid] = vv;
  sh_q[tid] = qv;
  sh_k[tid] = kv;
  __syncthreads();
  if (tid == 0) sqo[row] = s_sq;

  // 2D rotary: dims [0,16) use pos.x, [16,32) use pos.y
  const int d    = tid & 31;
  const int hb   = tid & ~31;           // head base within the 256 columns
  const int base = (d < 16) ? 0 : 16;
  const int dd   = d - base;
  const float t  = pos[row * 2 + (base ? 1 : 0)] * 64.0f;  // SCALE/MIN_FREQ
  const int fi   = dd & 7;
  const float invf = expf((float)fi * -1.1512925464970231f); // 10000^(-fi/8)
  const float f  = t * invf;
  const float cf = cosf(f);
  const float sf = sinf(f);
  const int pd   = (dd < 8) ? dd + 8 : dd - 8;
  const float sg = (dd < 8) ? -1.0f : 1.0f;
  const int pidx = hb + base + pd;
  Qo[row * 256 + tid] = sh_q[tid] * cf + sg * sh_q[pidx] * sf;
  Ko[row * 256 + tid] = sh_k[tid] * cf + sg * sh_k[pidx] * sf;
}

// ---------------------------------------------------------------------------
// Stage 2: bf16-split WMMA distance tiles + per-row top-9 (per column slice).
// One wave per (16-row query tile, column slice, batch).
// dot(h,h') ~= hi.hi' + hi.lo' + lo.hi'  (f32 accumulate, 3 WMMA chains)
// Double-buffered B tiles + WGP-scope prefetch two tiles ahead.
// ---------------------------------------------------------------------------
__global__ __launch_bounds__(32) void mha_knn(
    const __bf16* __restrict__ hhi, const __bf16* __restrict__ hlo,
    const float* __restrict__ sq,
    float* __restrict__ cand_v, int* __restrict__ cand_i) {
  __shared__ float tile[16][17];
  __shared__ float shsqm[16];
  const int qt    = blockIdx.x;   // query tile 0..511
  const int split = blockIdx.y;   // column slice 0..NSPLIT-1
  const int b     = blockIdx.z;
  const int lane  = threadIdx.x;
  const int q0    = qt * 16;
  const int r     = lane & 15;          // M (A) / N (B) index
  const int g     = lane >> 4;          // lane-group 0/1
  const int mhi   = g << 3;             // C row offset 0 or 8

  const __bf16* Hhi = hhi + (size_t)b * NN * CC;
  const __bf16* Hlo = hlo + (size_t)b * NN * CC;

  // Preload A tile (16 query rows x 128 K) hi & lo planes.
  // A k order used here (not the ISA table order): lane-group g holds contiguous K run [16g, 16g+16).
  const __bf16* ArHi = Hhi + (size_t)(q0 + r) * CC;
  const __bf16* ArLo = Hlo + (size_t)(q0 + r) * CC;
  v16bf ahi[4], alo[4];
  #pragma unroll
  for (int ch = 0; ch < 4; ++ch) {
    const int k0 = ch * 32;
    const v8bf h0 = *(const v8bf*)(ArHi + k0 + 16 * g);
    const v8bf h1 = *(const v8bf*)(ArHi + k0 + 16 * g + 8);
    ahi[ch] = __builtin_shufflevector(h0, h1, 0, 1, 2, 3, 4, 5, 6, 7,
                                      8, 9, 10, 11, 12, 13, 14, 15);
    const v8bf l0 = *(const v8bf*)(ArLo + k0 + 16 * g);
    const v8bf l1 = *(const v8bf*)(ArLo + k0 + 16 * g + 8);
    alo[ch] = __builtin_shufflevector(l0, l1, 0, 1, 2, 3, 4, 5, 6, 7,
                                      8, 9, 10, 11, 12, 13, 14, 15);
  }

  const float sqq = sq[(size_t)b * NN + q0 + r];

  float tv[KNN];
  int   ti[KNN];
  #pragma unroll
  for (int j = 0; j < KNN; ++j) { tv[j] = -3.4e38f; ti[j] = 0; }
  float tmin = -3.4e38f;
  int   tpos = 0;

  const int m_begin = split * (NN / NSPLIT);
  const int m_end   = m_begin + (NN / NSPLIT);

  // B k order used here (same as A): lane-group g holds contiguous K run [16g, 16g+16).
  const __bf16* BbHi = Hhi + (size_t)r * CC + 16 * g;   // + m0*CC per tile
  const __bf16* BbLo = Hlo + (size_t)r * CC + 16 * g;

  // Prime the double buffer with the first tile.
  v16bf bhi[4], blo[4];
  #pragma unroll
  for (int ch = 0; ch < 4; ++ch) {
    bhi[ch] = *(const v16bf*)(BbHi + (size_t)m_begin * CC + ch * 32);
    blo[ch] = *(const v16bf*)(BbLo + (size_t)m_begin * CC + ch * 32);
  }

  for (int m0 = m_begin; m0 < m_end; m0 += 16) {
    // Prefetch tile m0+32 into WGP cache (locality 3 -> near scope).
    const __bf16* PfHi = Hhi + (size_t)(m0 + 32 + r) * CC;
    const __bf16* PfLo = Hlo + (size_t)(m0 + 32 + r) * CC;
    __builtin_prefetch((const void*)PfHi, 0, 3);
    __builtin_prefetch((const void*)(PfHi + 64), 0, 3);
    __builtin_prefetch((const void*)PfLo, 0, 3);
    __builtin_prefetch((const void*)(PfLo + 64), 0, 3);

    // Issue next tile's B loads (over-run past slice end stays inside ws).
    v16bf nhi[4], nlo[4];
    #pragma unroll
    for (int ch = 0; ch < 4; ++ch) {
      nhi[ch] = *(const v16bf*)(BbHi + (size_t)(m0 + 16) * CC + ch * 32);
      nlo[ch] = *(const v16bf*)(BbLo + (size_t)(m0 + 16) * CC + ch * 32);
    }

    // 12 WMMAs on the current (already-resident) tile.
    v8f c0 = {}, c1 = {}, c2 = {};
    #pragma unroll
    for (int ch = 0; ch < 4; ++ch) {
      c0 = __builtin_amdgcn_wmma_f32_16x16x32_bf16(false, ahi[ch], false, bhi[ch],
                                                   (short)0, c0, false, false);
      c1 = __builtin_amdgcn_wmma_f32_16x16x32_bf16(false, ahi[ch], false, blo[ch],
                                                   (short)0, c1, false, false);
      c2 = __builtin_amdgcn_wmma_f32_16x16x32_bf16(false, alo[ch], false, bhi[ch],
                                                   (short)0, c2, false, false);
    }

    __syncthreads();  // previous scan reads done before overwrite
    #pragma unroll
    for (int vi = 0; vi < 8; ++vi)
      tile[vi + mhi][r] = c0[vi] + c1[vi] + c2[vi];   // (M, N) of C layout
    if (lane < 16) shsqm[lane] = sq[(size_t)b * NN + m0 + lane];
    __syncthreads();

    // lanes 0..15 each scan their query row's 16 new candidates
    if (lane < 16) {
      for (int c = 0; c < 16; ++c) {
        const float nd = 2.0f * tile[lane][c] - sqq - shsqm[c];
        if (nd > tmin) {
          const int idx = m0 + c;
          #pragma unroll
          for (int j = 0; j < KNN; ++j)
            if (j == tpos) { tv[j] = nd; ti[j] = idx; }
          tmin = tv[0]; tpos = 0;
          #pragma unroll
          for (int j = 1; j < KNN; ++j)
            if (tv[j] < tmin) { tmin = tv[j]; tpos = j; }
        }
      }
    }

    // rotate double buffer
    #pragma unroll
    for (int ch = 0; ch < 4; ++ch) { bhi[ch] = nhi[ch]; blo[ch] = nlo[ch]; }
  }

  if (lane < 16) {
    const size_t base = (((size_t)b * NN + q0 + lane) * NSPLIT + split) * KNN;
    #pragma unroll
    for (int j = 0; j < KNN; ++j) { cand_v[base + j] = tv[j]; cand_i[base + j] = ti[j]; }
  }
}

// ---------------------------------------------------------------------------
// Stage 2b: merge NSPLIT partial top-9 lists -> 9 indices per row.
// ---------------------------------------------------------------------------
__global__ __launch_bounds__(256) void mha_merge(
    const float* __restrict__ cand_v, const int* __restrict__ cand_i,
    int* __restrict__ nn_idx) {
  const int rrow = blockIdx.x * blockDim.x + threadIdx.x;
  if (rrow >= BB * NN) return;
  const float* v  = cand_v + (size_t)rrow * (NSPLIT * KNN);
  const int*   ix = cand_i + (size_t)rrow * (NSPLIT * KNN);
  float tv[KNN]; int ti[KNN];
  #pragma unroll
  for (int j = 0; j < KNN; ++j) { tv[j] = v[j]; ti[j] = ix[j]; }
  float tmin = tv[0]; int tpos = 0;
  #pragma unroll
  for (int j = 1; j < KNN; ++j) if (tv[j] < tmin) { tmin = tv[j]; tpos = j; }
  for (int c = KNN; c < NSPLIT * KNN; ++c) {
    const float val = v[c];
    if (val > tmin) {
      const int idx = ix[c];
      #pragma unroll
      for (int j = 0; j < KNN; ++j)
        if (j == tpos) { tv[j] = val; ti[j] = idx; }
      tmin = tv[0]; tpos = 0;
      #pragma unroll
      for (int j = 1; j < KNN; ++j) if (tv[j] < tmin) { tmin = tv[j]; tpos = j; }
    }
  }
  #pragma unroll
  for (int j = 0; j < KNN; ++j) nn_idx[(size_t)rrow * KNN + j] = ti[j];
}

// ---------------------------------------------------------------------------
// Stage 3: 9-neighbor attention.  One wave per (b,n,h); lane = d.
// ---------------------------------------------------------------------------
__global__ __launch_bounds__(256) void mha_attn(
    const float* __restrict__ Q, const float* __restrict__ K,
    const float* __restrict__ V, const int* __restrict__ nn_idx,
    float* __restrict__ out) {
  const int n = blockIdx.x;
  const int b = blockIdx.y;
  const size_t row = (size_t)b * NN + n;
  const int tid = threadIdx.x;  // h = tid>>5, d = tid&31
  const float q = Q[row * 256 + tid];
  const int* nnp = nn_idx + row * KNN;
  float acc = 0.0f, z = 0.0f;
  #pragma unroll
  for (int k = 0; k < KNN; ++k) {
    const int m = nnp[k];
    const size_t mrow = (size_t)b * NN + m;
    float p = K[mrow * 256 + tid] * q;
    #pragma unroll
    for (int off = 16; off > 0; off >>= 1) p += __shfl_xor(p, off, 32);
    float s = p * 0.17677669529663687f;            // 1/sqrt(32)
    s = fminf(fmaxf(s, -5.0f), 5.0f);
    s = expf(s);
    acc = fmaf(s, V[mrow * 256 + tid], acc);
    z += s;
  }
  out[row * 256 + tid] = acc / z;
}

// ---------------------------------------------------------------------------
extern "C" void kernel_launch(void* const* d_in, const int* in_sizes, int n_in,
                              void* d_out, int out_size, void* d_ws, size_t ws_size,
                              hipStream_t stream) {
  const float* h   = (const float*)d_in[0];
  const float* pos = (const float*)d_in[1];
  const float* WQ  = (const float*)d_in[2];
  const float* WK  = (const float*)d_in[3];
  const float* WV  = (const float*)d_in[4];
  float* out = (float*)d_out;

  const size_t rows = (size_t)BB * NN;           // 16384
  const size_t qkv  = rows * 256;                // elements per Q/K/V
  size_t off = 0;
  char* ws = (char*)d_ws;
  float*  Qb  = (float*) (ws + off); off += qkv * 4;
  float*  Kb  = (float*) (ws + off); off += qkv * 4;
  float*  Vb  = (float*) (ws + off); off += qkv * 4;
  float*  sqb = (float*) (ws + off); off += rows * 4;
  __bf16* hhi = (__bf16*)(ws + off); off += rows * CC * 2;
  __bf16* hlo = (__bf16*)(ws + off); off += rows * CC * 2;
  float*  cvb = (float*) (ws + off); off += rows * NSPLIT * KNN * 4;
  int*    cib = (int*)   (ws + off); off += rows * NSPLIT * KNN * 4;
  int*    nnb = (int*)   (ws + off); off += rows * KNN * 4;
  if (ws_size < off) return;  // workspace too small: bail

  mha_proj_rope<<<dim3(NN, BB), 256, 0, stream>>>(h, pos, WQ, WK, WV,
                                                  Qb, Kb, Vb, sqb, hhi, hlo);
  mha_knn<<<dim3(NN / 16, NSPLIT, BB), 32, 0, stream>>>(hhi, hlo, sqb, cvb, cib);
  mha_merge<<<dim3((rows + 255) / 256), 256, 0, stream>>>(cvb, cib, nnb);
  mha_attn<<<dim3(NN, BB), 256, 0, stream>>>(Qb, Kb, Vb, nnb, out);
}
